// SelfAttention_20186346291972
// MI455X (gfx1250) — matrix operand order repaired and checked by reading, not compiled, not run on hardware
//
#include <hip/hip_runtime.h>

typedef __attribute__((ext_vector_type(16))) __bf16 v16bf;
typedef __attribute__((ext_vector_type(8)))  __bf16 v8bf;
typedef __attribute__((ext_vector_type(8)))  float  v8f;

#define SEQ  2048
#define HD   1024
#define NB   8
#define MTOT (NB * SEQ)   // 16384 rows of x flattened

union AB { v16bf v; v8bf h[2]; };

#define WMMA_BF16(a, b, c) \
    __builtin_amdgcn_wmma_f32_16x16x32_bf16(false, (a), false, (b), (short)0, (c), false, false)

// ---------------------------------------------------------------------------
// Kernel 0a: one-time x f32 -> bf16 conversion (vectorized, memory bound).
// ---------------------------------------------------------------------------
__global__ __launch_bounds__(256) void cvt_x_kernel(
    const float* __restrict__ x, __bf16* __restrict__ xbf)
{
    size_t i = ((size_t)blockIdx.x * 256 + threadIdx.x) * 8;
    float4 a = *(const float4*)&x[i];
    float4 b = *(const float4*)&x[i + 4];
    union { v8bf v; __bf16 e[8]; } o;
    o.e[0] = (__bf16)a.x; o.e[1] = (__bf16)a.y;
    o.e[2] = (__bf16)a.z; o.e[3] = (__bf16)a.w;
    o.e[4] = (__bf16)b.x; o.e[5] = (__bf16)b.y;
    o.e[6] = (__bf16)b.z; o.e[7] = (__bf16)b.w;
    *(v8bf*)&xbf[i] = o.v;
}

// ---------------------------------------------------------------------------
// Kernel 0b: one-time weight transpose + f32->bf16:  Wt[m][n][k] = W_m[k][n].
// 32x32 LDS-tiled so both global sides stay coalesced.
// ---------------------------------------------------------------------------
__global__ __launch_bounds__(256) void cvt_w_kernel(
    const float* __restrict__ Wq, const float* __restrict__ Wk,
    const float* __restrict__ Wv, __bf16* __restrict__ Wt)
{
    __shared__ __bf16 t[32][33];
    const float* W = (blockIdx.z == 0) ? Wq : (blockIdx.z == 1) ? Wk : Wv;
    const int k0 = blockIdx.x * 32, n0 = blockIdx.y * 32;
    const int c = threadIdx.x & 31, r = threadIdx.x >> 5;   // r: 0..7
    #pragma unroll
    for (int i = 0; i < 4; ++i) {
        int kk = r + 8 * i;
        t[c][kk] = (__bf16)W[(size_t)(k0 + kk) * HD + n0 + c];   // coalesced in n
    }
    __syncthreads();
    __bf16* out = Wt + (size_t)blockIdx.z * HD * HD;
    #pragma unroll
    for (int i = 0; i < 4; ++i) {
        int nn = r + 8 * i;
        out[(size_t)(n0 + nn) * HD + k0 + c] = t[nn][c];         // coalesced in k
    }
}

// ---------------------------------------------------------------------------
// Kernel 1: fused QKV projection, no LDS: A/B fragments load straight from
// L2-resident bf16 x and pre-transposed weights as contiguous b128s.
// Block 128 thr (4 waves), tile 64 rows x 64 cols, all three matrices.
// Q pre-scaled by 1/sqrt(H); V written TRANSPOSED [B][H][S].
// ---------------------------------------------------------------------------
__global__ __launch_bounds__(128) void qkv_proj_kernel(
    const __bf16* __restrict__ xbf, const __bf16* __restrict__ Wt,
    const float* __restrict__ bq, const float* __restrict__ bk,
    const float* __restrict__ bv,
    __bf16* __restrict__ Qo, __bf16* __restrict__ Ko, __bf16* __restrict__ Vt)
{
    const int tid = threadIdx.x, lane = tid & 31, wv = tid >> 5;
    const int l15 = lane & 15, hi = (lane >> 4) & 1;
    const int m0 = blockIdx.x * 64, n0 = blockIdx.y * 64;
    const int alo = hi ? 16 : 0;

    v8f acc[3][4];
    #pragma unroll
    for (int m = 0; m < 3; ++m)
        #pragma unroll
        for (int c = 0; c < 4; ++c) acc[m][c] = (v8f){};

    const size_t arow = (size_t)(m0 + wv * 16 + l15) * HD;

    #pragma unroll 2
    for (int k0 = 0; k0 < HD; k0 += 32) {
        AB a;
        a.h[0] = *(const v8bf*)&xbf[arow + k0 + alo];
        a.h[1] = *(const v8bf*)&xbf[arow + k0 + alo + 8];
        #pragma unroll
        for (int m = 0; m < 3; ++m) {
            const __bf16* Wm = Wt + (size_t)m * HD * HD;
            #pragma unroll
            for (int c = 0; c < 4; ++c) {
                v16bf bfr = *(const v16bf*)
                    &Wm[(size_t)(n0 + c * 16 + l15) * HD + k0 + (hi ? 16 : 0)];
                acc[m][c] = WMMA_BF16(a.v, bfr, acc[m][c]);
            }
        }
    }

    const float qscale = 0.03125f;              // 1/sqrt(1024)
    const int   rbase  = hi ? 8 : 0;
    const float* Bv[3] = {bq, bk, bv};
    #pragma unroll
    for (int m = 0; m < 3; ++m) {
        #pragma unroll
        for (int c = 0; c < 4; ++c) {
            int n = n0 + c * 16 + l15;
            float bias = Bv[m][n];
            #pragma unroll
            for (int r = 0; r < 8; ++r) {
                int mm = m0 + wv * 16 + rbase + r;
                float v = acc[m][c][r] + bias;
                if (m == 0) {
                    Qo[(size_t)mm * HD + n] = (__bf16)(v * qscale);
                } else if (m == 1) {
                    Ko[(size_t)mm * HD + n] = (__bf16)v;
                } else {
                    int bb = mm >> 11, ss = mm & (SEQ - 1);
                    Vt[((size_t)bb * HD + n) * SEQ + ss] = (__bf16)v;
                }
            }
        }
    }
}

// ---------------------------------------------------------------------------
// Kernel 2: flash attention, kv tile = 64. Block 256 thr (8 waves), one
// 16-row q tile; wave w owns H-slice [128w,128w+128), O accum in registers.
// Partial scores reduced across waves via column-major LDS (b128 traffic).
// ---------------------------------------------------------------------------
__global__ __launch_bounds__(256) void attn_kernel(
    const __bf16* __restrict__ Q, const __bf16* __restrict__ K,
    const __bf16* __restrict__ Vt, float* __restrict__ out)
{
    __shared__ __align__(32) float  s_part[8][64][16];   // [wave][col][row] 32KB
    __shared__ __align__(32) float  s_red[64][16];       // 4KB
    __shared__ __align__(32) __bf16 p_lds[8][16][64];    // per-wave P buf 16KB

    const int tid = threadIdx.x, lane = tid & 31, wv = tid >> 5;
    const int b = blockIdx.y, q0 = blockIdx.x * 16;
    const int l15 = lane & 15, hi = (lane >> 4) & 1;
    const int rbase = hi ? 8 : 0, alo = hi ? 16 : 0;

    // Q A-fragments for this wave's 128-wide H slice, loaded once
    AB qa[4];
    {
        const size_t qrow = ((size_t)b * SEQ + q0 + l15) * HD;
        #pragma unroll
        for (int c = 0; c < 4; ++c) {
            int h0 = wv * 128 + c * 32;
            qa[c].h[0] = *(const v8bf*)&Q[qrow + h0 + alo];
            qa[c].h[1] = *(const v8bf*)&Q[qrow + h0 + alo + 8];
        }
    }

    v8f o_acc[8];
    #pragma unroll
    for (int c = 0; c < 8; ++c) o_acc[c] = (v8f){};
    float mrow[8], lrow[8];
    #pragma unroll
    for (int r = 0; r < 8; ++r) { mrow[r] = -__builtin_inff(); lrow[r] = 0.f; }

    const size_t kbase = (size_t)b * SEQ * HD;
    const size_t vtb   = (size_t)b * HD * SEQ;

    for (int kt = 0; kt < SEQ; kt += 64) {
        if (kt + 64 < SEQ)
            __builtin_prefetch(&K[kbase + (size_t)(kt + 64 + l15) * HD + wv * 128], 0, 1);

        // ---- partial scores S[16x64] over this wave's H slice ----
        v8f s[4];
        #pragma unroll
        for (int j = 0; j < 4; ++j) s[j] = (v8f){};
        #pragma unroll
        for (int c = 0; c < 4; ++c) {
            int kf = wv * 128 + c * 32 + (hi ? 16 : 0);
            #pragma unroll
            for (int j = 0; j < 4; ++j) {
                v16bf kb = *(const v16bf*)
                    &K[kbase + (size_t)(kt + j * 16 + l15) * HD + kf];
                s[j] = WMMA_BF16(qa[c].v, kb, s[j]);
            }
        }

        // ---- cross-wave reduction (column-major => contiguous v8f) ----
        #pragma unroll
        for (int j = 0; j < 4; ++j)
            *(v8f*)&s_part[wv][j * 16 + l15][rbase] = s[j];
        __syncthreads();
        #pragma unroll
        for (int e = 0; e < 2; ++e) {
            int idx = tid + 256 * e;            // 512 float2 segments
            int cc = idx >> 3, rp = (idx & 7) * 2;
            float sx = 0.f, sy = 0.f;
            #pragma unroll
            for (int w = 0; w < 8; ++w) {
                float2 v = *(const float2*)&s_part[w][cc][rp];
                sx += v.x; sy += v.y;
            }
            float2 o2; o2.x = sx; o2.y = sy;
            *(float2*)&s_red[cc][rp] = o2;
        }
        __syncthreads();
        #pragma unroll
        for (int j = 0; j < 4; ++j)
            s[j] = *(const v8f*)&s_red[j * 16 + l15][rbase];

        // ---- online softmax (rows in VGPR index, cols in 16-lane groups) ----
        #pragma unroll
        for (int r = 0; r < 8; ++r) {
            float t = fmaxf(fmaxf(s[0][r], s[1][r]), fmaxf(s[2][r], s[3][r]));
            #pragma unroll
            for (int off = 8; off; off >>= 1)
                t = fmaxf(t, __shfl_xor(t, off, 16));
            float mnew = fmaxf(mrow[r], t);
            float sc   = __expf(mrow[r] - mnew);
            float p0 = __expf(s[0][r] - mnew);
            float p1 = __expf(s[1][r] - mnew);
            float p2 = __expf(s[2][r] - mnew);
            float p3 = __expf(s[3][r] - mnew);
            float rs = (p0 + p1) + (p2 + p3);
            #pragma unroll
            for (int off = 8; off; off >>= 1)
                rs += __shfl_xor(rs, off, 16);
            lrow[r] = lrow[r] * sc + rs;
            mrow[r] = mnew;
            #pragma unroll
            for (int c = 0; c < 8; ++c) o_acc[c][r] *= sc;
            p_lds[wv][rbase + r][l15]      = (__bf16)p0;
            p_lds[wv][rbase + r][16 + l15] = (__bf16)p1;
            p_lds[wv][rbase + r][32 + l15] = (__bf16)p2;
            p_lds[wv][rbase + r][48 + l15] = (__bf16)p3;
        }
        asm volatile("s_wait_dscnt 0x0" ::: "memory");  // P visible within wave

        // ---- P A-fragments (16x64 -> two 16x32) ----
        AB pa0, pa1;
        pa0.h[0] = *(const v8bf*)&p_lds[wv][l15][alo];
        pa0.h[1] = *(const v8bf*)&p_lds[wv][l15][alo + 8];
        pa1.h[0] = *(const v8bf*)&p_lds[wv][l15][32 + alo];
        pa1.h[1] = *(const v8bf*)&p_lds[wv][l15][32 + alo + 8];

        // ---- O += P @ V  (V transposed -> contiguous B fragments) ----
        #pragma unroll
        for (int c = 0; c < 8; ++c) {
            size_t vrow = vtb + (size_t)(wv * 128 + c * 16 + l15) * SEQ;
            v16bf vb0 = *(const v16bf*)&Vt[vrow + kt +      (hi ? 16 : 0)];
            v16bf vb1 = *(const v16bf*)&Vt[vrow + kt + 32 + (hi ? 16 : 0)];
            o_acc[c] = WMMA_BF16(pa0.v, vb0, o_acc[c]);
            o_acc[c] = WMMA_BF16(pa1.v, vb1, o_acc[c]);
        }
    }

    // ---- normalize and write fp32 output ----
    #pragma unroll
    for (int r = 0; r < 8; ++r) {
        float inv = 1.f / lrow[r];
        size_t rowoff = ((size_t)b * SEQ + q0 + rbase + r) * HD;
        #pragma unroll
        for (int c = 0; c < 8; ++c)
            out[rowoff + wv * 128 + c * 16 + l15] = o_acc[c][r] * inv;
    }
}

// ---------------------------------------------------------------------------
extern "C" void kernel_launch(void* const* d_in, const int* in_sizes, int n_in,
                              void* d_out, int out_size, void* d_ws, size_t ws_size,
                              hipStream_t stream) {
    const float* x  = (const float*)d_in[0];
    const float* Wq = (const float*)d_in[1];
    const float* bq = (const float*)d_in[2];
    const float* Wk = (const float*)d_in[3];
    const float* bk = (const float*)d_in[4];
    const float* Wv = (const float*)d_in[5];
    const float* bv = (const float*)d_in[6];
    float* out = (float*)d_out;

    // Workspace: xbf 32MB | Wt 6MB | Q 32MB | K 32MB | V^T 32MB (all bf16)
    __bf16* xbf = (__bf16*)d_ws;
    __bf16* Wtw = xbf + (size_t)MTOT * HD;
    __bf16* Qw  = Wtw + (size_t)3 * HD * HD;
    __bf16* Kw  = Qw + (size_t)MTOT * HD;
    __bf16* Vt  = Kw + (size_t)MTOT * HD;

    cvt_x_kernel<<<(MTOT * HD) / (256 * 8), 256, 0, stream>>>(x, xbf);
    cvt_w_kernel<<<dim3(HD / 32, HD / 32, 3), 256, 0, stream>>>(Wq, Wk, Wv, Wtw);
    qkv_proj_kernel<<<dim3(MTOT / 64, HD / 64), 128, 0, stream>>>(
        xbf, Wtw, bq, bk, bv, Qw, Kw, Vt);
    attn_kernel<<<dim3(SEQ / 16, NB), 256, 0, stream>>>(Qw, Kw, Vt, out);
}
